// SNN_pMNIST_27513560498768
// MI455X (gfx1250) — matrix operand order repaired and checked by reading, not compiled, not run on hardware
//
#include <hip/hip_runtime.h>
#include <hip/hip_bf16.h>

// ---------------------------------------------------------------------------
// SNN pMNIST forward:  encode+LIF -> WMMA GEMM1 -> gate/BN -> LIF ->
//                      WMMA GEMM2 -> gate/BN -> readout GEMM -> mean over T
// B=512, T=10, D0=784(pad 800), D1=D2=4096, D3=10, M = B*T = 5120
// GEMMs: v_wmma_f32_16x16x32_f16 with double-buffered
//        global_load_async_to_lds_b128 staging (ASYNCcnt pipeline).
// ---------------------------------------------------------------------------

typedef __attribute__((ext_vector_type(16))) _Float16 v16h;
typedef __attribute__((ext_vector_type(8)))  float    v8f;

#define TAU   0.25f
#define V_TH  1.0f
#define BN_EPS 1e-5f

#define BATCH 512
#define TSTEP 10
#define MROWS (BATCH * TSTEP)   // 5120
#define D0    784
#define K0P   800               // 784 padded to multiple of 32
#define D1    4096
#define D2    4096
#define D3    10

// ---------------------------------------------------------------------------
// Kernel 1: Poisson encode + LIF layer 0. Output f16 spikes [5120, 800].
// ---------------------------------------------------------------------------
__global__ void k_encode_lif(const float* __restrict__ x,
                             const float* __restrict__ noise,
                             _Float16* __restrict__ spk) {
    int idx = blockIdx.x * blockDim.x + threadIdx.x;      // 512*800
    if (idx >= BATCH * K0P) return;
    int b = idx / K0P;
    int d = idx - b * K0P;
    if (d >= D0) {
        #pragma unroll
        for (int t = 0; t < TSTEP; ++t)
            spk[(size_t)(b * TSTEP + t) * K0P + d] = (_Float16)0.0f;
        return;
    }
    float xv = x[(size_t)b * D0 + d];
    float mem = 0.0f;
    #pragma unroll
    for (int t = 0; t < TSTEP; ++t) {
        float nv = noise[((size_t)b * TSTEP + t) * D0 + d];
        float h  = (nv <= xv) ? 1.0f : 0.0f;              // Poisson spike
        mem = TAU * mem + h;
        float s = ((mem - V_TH) > 0.0f) ? 1.0f : 0.0f;    // heaviside
        spk[(size_t)(b * TSTEP + t) * K0P + d] = (_Float16)s;
        mem *= (1.0f - s);
    }
}

// ---------------------------------------------------------------------------
// f32 -> f16 weight conversion (with optional K padding).
// ---------------------------------------------------------------------------
__global__ void k_cvt_w_pad(const float* __restrict__ W, _Float16* __restrict__ Wh,
                            int N, int K, int Kpad) {
    int idx = blockIdx.x * blockDim.x + threadIdx.x;
    if (idx >= N * Kpad) return;
    int n = idx / Kpad;
    int k = idx - n * Kpad;
    Wh[idx] = (k < K) ? (_Float16)W[(size_t)n * K + k] : (_Float16)0.0f;
}

__global__ void k_cvt_w(const float* __restrict__ W, _Float16* __restrict__ Wh, int n) {
    int idx = blockIdx.x * blockDim.x + threadIdx.x;
    if (idx < n) Wh[idx] = (_Float16)W[idx];
}

// ---------------------------------------------------------------------------
// Async-copy helpers (CDNA5 GLOBAL_LOAD_ASYNC_TO_LDS, tracked on ASYNCcnt).
// LDS flat addresses carry the byte offset in addr[31:0] (aperture rules),
// so truncating the generic pointer yields the dsaddr operand.
// ---------------------------------------------------------------------------
__device__ __forceinline__ unsigned lds_addr32(const void* p) {
    return (unsigned)(unsigned long long)(size_t)p;
}
__device__ __forceinline__ void async_ld_b128(unsigned lds_off, const void* g) {
    asm volatile("global_load_async_to_lds_b128 %0, %1, off"
                 :: "v"(lds_off), "v"((unsigned long long)(size_t)g)
                 : "memory");
}
__device__ __forceinline__ void wait_async0() {
    asm volatile("s_wait_asynccnt 0x0" ::: "memory");
}

// ---------------------------------------------------------------------------
// WMMA GEMM:  C[M,N] = A[M,K] * W[N,K]^T, epilogue (c + bias[n]) * gate[m/T][n]
// A row-major f16, W row-major f16 (B[k][n] = W[n][k]).
// Block tile 128x128, BK=32, 256 threads = 8 wave32 waves (2x4 wave grid),
// each wave computes 64x32 via 4x2 v_wmma_f32_16x16x32_f16 accumulators.
// Double-buffered LDS; tiles fetched with global_load_async_to_lds_b128.
// M,N,K multiples of 128/128/32 for all call sites.
// ---------------------------------------------------------------------------
#define BM 128
#define BN 128
#define BK 32
#define LDSP 40   // 40 halfs = 80B row stride: 16B-aligned rows, low-conflict

__global__ __launch_bounds__(256)
void k_gemm_wmma(const _Float16* __restrict__ A,
                 const _Float16* __restrict__ W,
                 const float* __restrict__ bias,
                 const float* __restrict__ gate,
                 float* __restrict__ C,
                 int M, int N, int K) {
    __shared__ _Float16 As[2][BM][LDSP];
    __shared__ _Float16 Bs[2][BN][LDSP];

    const int tid  = threadIdx.x;
    const int lane = tid & 31;
    const int wave = tid >> 5;          // 0..7
    const int wm   = wave >> 2;         // 0..1 -> 64-row slab
    const int wn   = wave & 3;          // 0..3 -> 32-col slab
    const int r    = lane & 15;
    const int hi   = lane >> 4;         // 0 or 1

    const int blockN = blockIdx.x * BN;
    const int blockM = blockIdx.y * BM;

    // staging coordinates: 512 chunks of 8 halfs (16B) per tile, 2 per thread
    const int row0 = tid >> 2;                 // chunk l = tid
    const int cc0  = (tid & 3) * 8;
    const int row1 = (tid + 256) >> 2;         // chunk l = tid + 256
    const int cc1  = ((tid + 256) & 3) * 8;

    v8f acc[4][2];
    #pragma unroll
    for (int i = 0; i < 4; ++i)
        #pragma unroll
        for (int j = 0; j < 2; ++j)
            #pragma unroll
            for (int q = 0; q < 8; ++q)
                acc[i][j][q] = 0.0f;

    const int ntiles = K / BK;

    // prologue: async-fetch tile 0 into buffer 0
    {
        const int kk = 0;
        async_ld_b128(lds_addr32(&As[0][row0][cc0]),
                      A + (size_t)(blockM + row0) * K + kk + cc0);
        async_ld_b128(lds_addr32(&As[0][row1][cc1]),
                      A + (size_t)(blockM + row1) * K + kk + cc1);
        async_ld_b128(lds_addr32(&Bs[0][row0][cc0]),
                      W + (size_t)(blockN + row0) * K + kk + cc0);
        async_ld_b128(lds_addr32(&Bs[0][row1][cc1]),
                      W + (size_t)(blockN + row1) * K + kk + cc1);
    }
    wait_async0();
    __syncthreads();

    for (int ki = 0; ki < ntiles; ++ki) {
        const int cur = ki & 1;
        const int nxt = cur ^ 1;

        // prefetch next K-tile into the alternate buffer (overlaps WMMA)
        if (ki + 1 < ntiles) {
            const int kk = (ki + 1) * BK;
            async_ld_b128(lds_addr32(&As[nxt][row0][cc0]),
                          A + (size_t)(blockM + row0) * K + kk + cc0);
            async_ld_b128(lds_addr32(&As[nxt][row1][cc1]),
                          A + (size_t)(blockM + row1) * K + kk + cc1);
            async_ld_b128(lds_addr32(&Bs[nxt][row0][cc0]),
                          W + (size_t)(blockN + row0) * K + kk + cc0);
            async_ld_b128(lds_addr32(&Bs[nxt][row1][cc1]),
                          W + (size_t)(blockN + row1) * K + kk + cc1);
        }

        // ---- load fragments per ISA 7.12.2 layouts ----
        v16h afrag[4];
        #pragma unroll
        for (int i = 0; i < 4; ++i) {
            const int arow = wm * 64 + i * 16 + r;
            #pragma unroll
            for (int e = 0; e < 8; ++e) {
                // A 16x32 f16: lanes0-15 K in {0..7,16..23}, lanes16-31 +8
                int k = ((e & 3) * 2) + ((e >> 2) * 16) + hi * 8;
                afrag[i][2 * e]     = As[cur][arow][k];
                afrag[i][2 * e + 1] = As[cur][arow][k + 1];
            }
        }
        v16h bfrag[2];
        #pragma unroll
        for (int j = 0; j < 2; ++j) {
            const int bcol = wn * 32 + j * 16 + r;
            #pragma unroll
            for (int e = 0; e < 8; ++e) {
                // B 32x16 f16: lanes0-15 K in {0..7,16..23}, lanes16-31 +8
                int k = ((e & 3) * 2) + ((e >> 2) * 16) + hi * 8;
                bfrag[j][2 * e]     = Bs[cur][bcol][k];
                bfrag[j][2 * e + 1] = Bs[cur][bcol][k + 1];
            }
        }

        #pragma unroll
        for (int i = 0; i < 4; ++i)
            #pragma unroll
            for (int j = 0; j < 2; ++j)
                acc[i][j] = __builtin_amdgcn_wmma_f32_16x16x32_f16(
                    false, afrag[i], false, bfrag[j],
                    (short)0, acc[i][j], false, false);

        // prefetch must have landed + all waves done reading cur before swap
        wait_async0();
        __syncthreads();
    }

    // ---- epilogue: (acc + bias) * gate[row/T], store f32 ----
    #pragma unroll
    for (int j = 0; j < 2; ++j) {
        const int gn = blockN + wn * 32 + j * 16 + r;
        const float bv = bias[gn];
        #pragma unroll
        for (int i = 0; i < 4; ++i) {
            #pragma unroll
            for (int q = 0; q < 8; ++q) {
                // C 16x16 f32: VGPR q -> M=q (lanes0-15) / M=q+8 (lanes16-31)
                unsigned gm = (unsigned)(blockM + wm * 64 + i * 16 + hi * 8 + q);
                float g = gate[(size_t)(gm / (unsigned)TSTEP) * N + gn];
                C[(size_t)gm * N + gn] = (acc[i][j][q] + bv) * g;
            }
        }
    }
}

// ---------------------------------------------------------------------------
// BN stats, deterministic two-phase column reduction over R=5120 rows.
// ---------------------------------------------------------------------------
#define BN_CHUNKS 16
__global__ void k_bn_partial(const float* __restrict__ P,
                             float* __restrict__ psum, float* __restrict__ psq,
                             int R, int C) {
    int c = blockIdx.x * blockDim.x + threadIdx.x;
    if (c >= C) return;
    int rows = (R + BN_CHUNKS - 1) / BN_CHUNKS;
    int r0 = blockIdx.y * rows;
    int r1 = r0 + rows; if (r1 > R) r1 = R;
    float s = 0.0f, sq = 0.0f;
    for (int r = r0; r < r1; ++r) {
        float v = P[(size_t)r * C + c];
        s += v; sq += v * v;
    }
    psum[(size_t)blockIdx.y * C + c] = s;
    psq [(size_t)blockIdx.y * C + c] = sq;
}

__global__ void k_bn_finalize(const float* __restrict__ psum,
                              const float* __restrict__ psq,
                              float* __restrict__ mean, float* __restrict__ rstd,
                              int R, int C) {
    int c = blockIdx.x * blockDim.x + threadIdx.x;
    if (c >= C) return;
    float s = 0.0f, sq = 0.0f;
    #pragma unroll
    for (int ch = 0; ch < BN_CHUNKS; ++ch) {
        s  += psum[(size_t)ch * C + c];
        sq += psq [(size_t)ch * C + c];
    }
    float mu  = s / (float)R;
    float var = sq / (float)R - mu * mu;
    mean[c] = mu;
    rstd[c] = rsqrtf(var + BN_EPS);
}

// ---------------------------------------------------------------------------
// BN apply + LIF (sequential over T) -> next-layer f16 spikes.
// ---------------------------------------------------------------------------
__global__ void k_bn_lif(const float* __restrict__ P,
                         const float* __restrict__ mean, const float* __restrict__ rstd,
                         const float* __restrict__ bnw, const float* __restrict__ bnb,
                         _Float16* __restrict__ spk, int C) {
    int idx = blockIdx.x * blockDim.x + threadIdx.x;    // 512*4096
    if (idx >= BATCH * C) return;
    int b = idx / C;
    int f = idx - b * C;
    float mu = mean[f], rs = rstd[f], w = bnw[f], bb = bnb[f];
    float mem = 0.0f;
    #pragma unroll
    for (int t = 0; t < TSTEP; ++t) {
        size_t off = (size_t)(b * TSTEP + t) * C + f;
        float v = (P[off] - mu) * rs * w + bb;
        mem = TAU * mem + v;
        float s = ((mem - V_TH) > 0.0f) ? 1.0f : 0.0f;
        spk[off] = (_Float16)s;
        mem *= (1.0f - s);
    }
}

// ---------------------------------------------------------------------------
// BN apply + sum over T:  S[b][f] = sum_t bnorm(P[b*T+t][f]).
// ---------------------------------------------------------------------------
__global__ void k_bn_sum_t(const float* __restrict__ P,
                           const float* __restrict__ mean, const float* __restrict__ rstd,
                           const float* __restrict__ bnw, const float* __restrict__ bnb,
                           float* __restrict__ S, int C) {
    int idx = blockIdx.x * blockDim.x + threadIdx.x;
    if (idx >= BATCH * C) return;
    int b = idx / C;
    int f = idx - b * C;
    float mu = mean[f], rs = rstd[f], w = bnw[f], bb = bnb[f];
    float acc = 0.0f;
    #pragma unroll
    for (int t = 0; t < TSTEP; ++t)
        acc += (P[(size_t)(b * TSTEP + t) * C + f] - mu) * rs * w + bb;
    S[(size_t)b * C + f] = acc;
}

// ---------------------------------------------------------------------------
// Readout: out[b][k] = (1/T) * sum_o W3[k][o] * S[b][o] + b3[k].  Tiny GEMM.
// ---------------------------------------------------------------------------
__global__ void k_readout(const float* __restrict__ S,
                          const float* __restrict__ W3, const float* __restrict__ b3,
                          float* __restrict__ out) {
    int idx = blockIdx.x * blockDim.x + threadIdx.x;    // 512*10
    if (idx >= BATCH * D3) return;
    int b = idx / D3;
    int k = idx - b * D3;
    const float4* sp = (const float4*)(S  + (size_t)b * D2);
    const float4* wr = (const float4*)(W3 + (size_t)k * D2);
    float acc = 0.0f;
    for (int o = 0; o < D2 / 4; ++o) {
        float4 a = sp[o], w = wr[o];
        acc += a.x * w.x + a.y * w.y + a.z * w.z + a.w * w.w;
    }
    out[idx] = acc * (1.0f / (float)TSTEP) + b3[k];
}

// ---------------------------------------------------------------------------
// Launch
// ---------------------------------------------------------------------------
extern "C" void kernel_launch(void* const* d_in, const int* in_sizes, int n_in,
                              void* d_out, int out_size, void* d_ws, size_t ws_size,
                              hipStream_t stream) {
    const float* x     = (const float*)d_in[0];
    const float* noise = (const float*)d_in[1];
    const float* gate1 = (const float*)d_in[2];
    const float* gate2 = (const float*)d_in[3];
    const float* W1    = (const float*)d_in[4];
    const float* b1    = (const float*)d_in[5];
    const float* bnw1  = (const float*)d_in[6];
    const float* bnb1  = (const float*)d_in[7];
    const float* W2    = (const float*)d_in[8];
    const float* b2    = (const float*)d_in[9];
    const float* bnw2  = (const float*)d_in[10];
    const float* bnb2  = (const float*)d_in[11];
    const float* W3    = (const float*)d_in[12];
    const float* b3    = (const float*)d_in[13];
    float* out = (float*)d_out;

    // workspace carve-up (256B aligned)
    char* wp = (char*)d_ws;
    auto carve = [&](size_t bytes) {
        void* p = (void*)wp;
        wp += (bytes + 255) & ~(size_t)255;
        return p;
    };
    _Float16* spk1 = (_Float16*)carve((size_t)MROWS * K0P * sizeof(_Float16));
    _Float16* w1h  = (_Float16*)carve((size_t)D1 * K0P * sizeof(_Float16));
    float*    P    = (float*)   carve((size_t)MROWS * D1 * sizeof(float));     // reused for layer2
    _Float16* spk2 = (_Float16*)carve((size_t)MROWS * D1 * sizeof(_Float16));
    _Float16* w2h  = (_Float16*)carve((size_t)D2 * D1 * sizeof(_Float16));
    float*    psum = (float*)   carve((size_t)BN_CHUNKS * D1 * sizeof(float));
    float*    psq  = (float*)   carve((size_t)BN_CHUNKS * D1 * sizeof(float));
    float*    mean = (float*)   carve((size_t)D1 * sizeof(float));
    float*    rstd = (float*)   carve((size_t)D1 * sizeof(float));
    float*    S    = (float*)   carve((size_t)BATCH * D2 * sizeof(float));

    const int TB = 256;

    // 1. Poisson encode + LIF layer 0 (padded to K0P)
    k_encode_lif<<<(BATCH * K0P + TB - 1) / TB, TB, 0, stream>>>(x, noise, spk1);

    // 2. weight conversions
    k_cvt_w_pad<<<(D1 * K0P + TB - 1) / TB, TB, 0, stream>>>(W1, w1h, D1, D0, K0P);
    k_cvt_w<<<(D2 * D1 + TB - 1) / TB, TB, 0, stream>>>(W2, w2h, D2 * D1);

    // 3. GEMM1 (WMMA) + bias + gate1 -> P [5120, 4096]
    k_gemm_wmma<<<dim3(D1 / BN, MROWS / BM), TB, 0, stream>>>(
        spk1, w1h, b1, gate1, P, MROWS, D1, K0P);

    // 4. BN1 stats, then BN+LIF -> spk2
    k_bn_partial<<<dim3(D1 / TB, BN_CHUNKS), TB, 0, stream>>>(P, psum, psq, MROWS, D1);
    k_bn_finalize<<<D1 / TB, TB, 0, stream>>>(psum, psq, mean, rstd, MROWS, D1);
    k_bn_lif<<<(BATCH * D1 + TB - 1) / TB, TB, 0, stream>>>(
        P, mean, rstd, bnw1, bnb1, spk2, D1);

    // 5. GEMM2 (WMMA) + bias + gate2 -> P (reuse)
    k_gemm_wmma<<<dim3(D2 / BN, MROWS / BM), TB, 0, stream>>>(
        spk2, w2h, b2, gate2, P, MROWS, D2, D1);

    // 6. BN2 stats, then BN + sum over T -> S [512, 4096]
    k_bn_partial<<<dim3(D2 / TB, BN_CHUNKS), TB, 0, stream>>>(P, psum, psq, MROWS, D2);
    k_bn_finalize<<<D2 / TB, TB, 0, stream>>>(psum, psq, mean, rstd, MROWS, D2);
    k_bn_sum_t<<<(BATCH * D2 + TB - 1) / TB, TB, 0, stream>>>(
        P, mean, rstd, bnw2, bnb2, S, D2);

    // 7. readout + time mean -> out [512, 10]
    k_readout<<<(BATCH * D3 + TB - 1) / TB, TB, 0, stream>>>(S, W3, b3, out);
}
